// GraphCastProcessor_26585847562366
// MI455X (gfx1250) — matrix operand order repaired and checked by reading, not compiled, not run on hardware
//
#include <hip/hip_runtime.h>
#include <hip/hip_bf16.h>
#include <cstdint>

#define HDIM    512
#define NNODES  10242
#define NEDGES  81920
#define NLAYERS 4

#define BM 32
#define BN 512
#define BK 32

typedef __attribute__((ext_vector_type(16))) __bf16 bf16x16;
typedef __attribute__((ext_vector_type(8)))  __bf16 bf16x8;
typedef __attribute__((ext_vector_type(8)))  float  floatx8;
typedef __attribute__((ext_vector_type(4)))  float  fx4;
typedef __attribute__((ext_vector_type(2)))  unsigned int ux2;

// explicit global-address-space pointers: keep prefetch loads on the GLOBAL path
// (flat loads would tick both LOADcnt and DScnt and stall at issue)
typedef const fx4 __attribute__((address_space(1)))* gf4p;
typedef const ux2 __attribute__((address_space(1)))* gu2p;

union SMem {
  struct {
    __bf16 A[2][BM][BK + 8];   // double-buffered 32 x 40 bf16
    __bf16 B[2][BN][BK + 8];   // double-buffered 512 x 40 bf16 ([n][k] transposed)
  } st;
  struct {
    float p1[BM][8];
    float p2[BM][8];
    float mean[BM];
    float rstd[BM];
  } red;
};

// generic shared pointer -> 32-bit LDS offset (generic LDS addr truncates to [31:0])
__device__ __forceinline__ uint32_t lds_off(const void* p) {
  return (uint32_t)(uintptr_t)p;
}

// ASRC: 0 = plain bf16 A [M,K];  1 = edge gather concat(e, n[src], n[dst]);
//       2 = node concat(n, agg)
// EPI:  0 = bias + SiLU -> bf16;  1 = bias + LayerNorm + residual -> f32 (in place)
template <int ASRC, int EPI, int K>
__global__ __launch_bounds__(256) void gemm_fused(
    const __bf16* __restrict__ Aplain,
    const float*  __restrict__ g0,
    const float*  __restrict__ g1,
    const float*  __restrict__ g2,
    const int*    __restrict__ idx1,
    const int*    __restrict__ idx2,
    const __bf16* __restrict__ Bt,   // [BN][K] bf16, transposed weights
    int M,
    const float*  __restrict__ bias,
    __bf16*       __restrict__ outb,
    float*        __restrict__ outf,
    const float*  __restrict__ lns,
    const float*  __restrict__ lnb)
{
  __shared__ SMem sm;

  const int tid  = threadIdx.x;
  const int wid  = tid >> 5;
  const int lane = tid & 31;
  const int lr   = lane & 15;
  const int hi   = lane >> 4;
  const int wn0  = wid * 64;              // this wave's N offset (4 N-tiles)
  const int m0   = blockIdx.x * BM;

  floatx8 acc[2][4];
#pragma unroll
  for (int mt = 0; mt < 2; ++mt)
#pragma unroll
    for (int nt = 0; nt < 4; ++nt)
#pragma unroll
      for (int i = 0; i < 8; ++i) acc[mt][nt][i] = 0.0f;

  // A-staging role of this thread: 8 threads per row, 4 elems each
  const int ar  = tid >> 3;
  const int akc = (tid & 7) * 4;
  int gr = m0 + ar;
  if (gr >= M) gr = M - 1;

  // ---- hoist per-row A base pointers (incl. gather index loads) out of k-loop ----
  constexpr int NSEG = (ASRC == 0) ? 1 : K / HDIM;
  constexpr int T    = K / BK;            // number of k-tiles
  const __bf16* pbase = nullptr;
  const float*  fbase[3] = {nullptr, nullptr, nullptr};
  if constexpr (ASRC == 0) {
    pbase = Aplain + (size_t)gr * K;
  } else if constexpr (ASRC == 1) {
    fbase[0] = g0 + (size_t)gr * HDIM;
    fbase[1] = g1 + (size_t)idx1[gr] * HDIM;
    fbase[2] = g2 + (size_t)idx2[gr] * HDIM;
  } else {
    fbase[0] = g0 + (size_t)gr * HDIM;
    fbase[1] = g1 + (size_t)gr * HDIM;
  }

  // load A-tile global data for tile t into registers (no LDS yet)
  auto ldA = [&](int t, fx4& vf, ux2& vu) {
    const int kt = t * BK;
    if constexpr (ASRC == 0) {
      vu = *(gu2p)(pbase + kt + akc);
    } else {
      const int seg = t >> 4;             // HDIM/BK == 16 tiles per segment
      const float* fb = fbase[0];
      if constexpr (NSEG > 1) { if (seg == 1) fb = fbase[1]; }
      if constexpr (NSEG > 2) { if (seg == 2) fb = fbase[2]; }
      vf = *(gf4p)(fb + (kt & (HDIM - 1)) + akc);
    }
  };
  // convert + store prefetched A data into LDS buffer t&1
  auto stA = [&](int t, const fx4& vf, const ux2& vu) {
    const int b = t & 1;
    if constexpr (ASRC == 0) {
      *(ux2*)&sm.st.A[b][ar][akc] = vu;
    } else {
      __bf16* d = &sm.st.A[b][ar][akc];
      d[0] = (__bf16)vf[0]; d[1] = (__bf16)vf[1];
      d[2] = (__bf16)vf[2]; d[3] = (__bf16)vf[3];
    }
  };
  // issue async global->LDS loads for B tile t (8 instructions per wave, ASYNCcnt)
  auto issueB = [&](int t) {
    const int kt = t * BK;
    const int b  = t & 1;
#pragma unroll
    for (int j = 0; j < 8; ++j) {
      const int idx = tid + j * 256;     // 0..2047 16B chunks
      const int n   = idx >> 2;
      const int c4  = idx & 3;
      const __bf16* gsrc = Bt + (size_t)n * K + kt + c4 * 8;
      const uint32_t laddr = lds_off(&sm.st.B[b][n][c4 * 8]);
      asm volatile("global_load_async_to_lds_b128 %0, %1, off"
                   :: "v"(laddr), "v"(gsrc)
                   : "memory");
    }
  };
  // 8 WMMAs from LDS buffer b
  auto compute = [&](int b) {
    // A layout (ISA 7.12.2): lanes 0-15 hold K {0..7,16..23}; 16-31 hold {8..15,24..31}
    bf16x16 af[2];
#pragma unroll
    for (int mt = 0; mt < 2; ++mt) {
      const __bf16* rp = &sm.st.A[b][mt * 16 + lr][0];
      const bf16x8 lo  = *(const bf16x8*)(rp + hi * 8);
      const bf16x8 hi8 = *(const bf16x8*)(rp + hi * 8 + 16);
#pragma unroll
      for (int i = 0; i < 8; ++i) { af[mt][i] = lo[i]; af[mt][i + 8] = hi8[i]; }
    }
    // B layout: lanes 0-15 (col = lr) hold K {0..7,16..23}; lanes 16-31 hold {8..15,24..31}
#pragma unroll
    for (int nt = 0; nt < 4; ++nt) {
      const __bf16* rp = &sm.st.B[b][wn0 + nt * 16 + lr][0];
      const bf16x8 lo  = *(const bf16x8*)(rp + hi * 8);
      const bf16x8 hi8 = *(const bf16x8*)(rp + hi * 8 + 16);
      bf16x16 bfrag;
#pragma unroll
      for (int i = 0; i < 8; ++i) { bfrag[i] = lo[i]; bfrag[i + 8] = hi8[i]; }

      acc[0][nt] = __builtin_amdgcn_wmma_f32_16x16x32_bf16(
          false, af[0], false, bfrag, (short)0, acc[0][nt], false, false);
      acc[1][nt] = __builtin_amdgcn_wmma_f32_16x16x32_bf16(
          false, af[1], false, bfrag, (short)0, acc[1][nt], false, false);
    }
  };

  // -------- software pipeline: B async double-buffered, A prefetched to VGPRs --------
  {
    fx4 vf = {0.0f, 0.0f, 0.0f, 0.0f}; ux2 vu = {0u, 0u};
    ldA(0, vf, vu); stA(0, vf, vu); issueB(0);
    ldA(1, vf, vu); stA(1, vf, vu); issueB(1);
  }
  fx4 pf = {0.0f, 0.0f, 0.0f, 0.0f}; ux2 pu = {0u, 0u};
#pragma unroll 1
  for (int t = 0; t < T; ++t) {
    // in-order ASYNCcnt: <=8 pending means tile t's 8 loads have completed
    if (t < T - 1) asm volatile("s_wait_asynccnt 0x8" ::: "memory");
    else           asm volatile("s_wait_asynccnt 0x0" ::: "memory");
    __syncthreads();                      // tile t visible to all waves
    const bool pre = (t + 2 < T);
    if (pre) ldA(t + 2, pf, pu);          // A prefetch (global path) overlaps compute
    compute(t & 1);
    __syncthreads();                      // all waves done reading buffer t&1
    if (pre) { stA(t + 2, pf, pu); issueB(t + 2); }
  }

  // ---------------- epilogue ----------------
  // C/D layout: VGPR r: lanes 0-15 -> M = r, lanes 16-31 -> M = r + 8; N = lr
  if constexpr (EPI == 0) {
#pragma unroll
    for (int nt = 0; nt < 4; ++nt) {
      const int col  = wn0 + nt * 16 + lr;
      const float bv = bias[col];
#pragma unroll
      for (int mt = 0; mt < 2; ++mt)
#pragma unroll
        for (int r8 = 0; r8 < 8; ++r8) {
          const int grow = m0 + mt * 16 + r8 + hi * 8;
          if (grow < M) {
            const float v = acc[mt][nt][r8] + bv;
            outb[(size_t)grow * HDIM + col] = (__bf16)(v / (1.0f + __expf(-v)));
          }
        }
    }
  } else {
    __syncthreads();   // done with st.* before reusing union as red.*

    // per-wave partial row sums over this wave's 64 columns
#pragma unroll
    for (int mt = 0; mt < 2; ++mt)
#pragma unroll
      for (int r8 = 0; r8 < 8; ++r8) {
        const int ml = mt * 16 + r8 + hi * 8;
        float s1 = 0.0f, s2 = 0.0f;
#pragma unroll
        for (int nt = 0; nt < 4; ++nt) {
          const float v = acc[mt][nt][r8] + bias[wn0 + nt * 16 + lr];
          s1 += v; s2 += v * v;
        }
#pragma unroll
        for (int off = 1; off < 16; off <<= 1) {  // reduce within 16-lane half
          s1 += __shfl_xor(s1, off, 32);
          s2 += __shfl_xor(s2, off, 32);
        }
        if (lr == 0) { sm.red.p1[ml][wid] = s1; sm.red.p2[ml][wid] = s2; }
      }
    __syncthreads();

    if (tid < BM) {
      float s1 = 0.0f, s2 = 0.0f;
#pragma unroll
      for (int w = 0; w < 8; ++w) { s1 += sm.red.p1[tid][w]; s2 += sm.red.p2[tid][w]; }
      const float mean = s1 * (1.0f / HDIM);
      const float var  = s2 * (1.0f / HDIM) - mean * mean;
      sm.red.mean[tid] = mean;
      sm.red.rstd[tid] = rsqrtf(var + 1e-5f);
    }
    __syncthreads();

#pragma unroll
    for (int nt = 0; nt < 4; ++nt) {
      const int col  = wn0 + nt * 16 + lr;
      const float bv = bias[col];
      const float sc = lns[col];
      const float bb = lnb[col];
#pragma unroll
      for (int mt = 0; mt < 2; ++mt)
#pragma unroll
        for (int r8 = 0; r8 < 8; ++r8) {
          const int ml   = mt * 16 + r8 + hi * 8;
          const int grow = m0 + ml;
          if (grow < M) {
            const float v = acc[mt][nt][r8] + bv;
            float* op = outf + (size_t)grow * HDIM + col;
            *op = *op + (v - sm.red.mean[ml]) * sm.red.rstd[ml] * sc + bb;
          }
        }
    }
  }
}

// W [K,N] f32 -> Wt [N,K] bf16 (run once per matrix; makes GEMM B staging contiguous)
__global__ void wtrans_kernel(const float* __restrict__ W, __bf16* __restrict__ Wt,
                              int K, int N) {
  const size_t idx = (size_t)blockIdx.x * 256 + threadIdx.x;
  if (idx >= (size_t)K * N) return;
  const int n = (int)(idx / K);
  const int k = (int)(idx % K);
  Wt[idx] = (__bf16)W[(size_t)k * N + n];
}

__global__ void zero_kernel(float* __restrict__ p, int n) {
  const int i = blockIdx.x * 256 + threadIdx.x;
  if (i < n) p[i] = 0.0f;
}

// segment_sum(e, dst): one block per edge, fp32 atomics land in L2 (agg = 21 MB, L2-resident)
__global__ void scatter_kernel(const float* __restrict__ e, const int* __restrict__ dst,
                               float* __restrict__ agg) {
  const int i = blockIdx.x;
  const int c = threadIdx.x;
  const int d = dst[i];
  const float* er = e + (size_t)i * HDIM;
  float* ar = agg + (size_t)d * HDIM;
  atomicAdd(&ar[c], er[c]);
  atomicAdd(&ar[c + 256], er[c + 256]);
}

extern "C" void kernel_launch(void* const* d_in, const int* in_sizes, int n_in,
                              void* d_out, int out_size, void* d_ws, size_t ws_size,
                              hipStream_t stream) {
  (void)in_sizes; (void)n_in; (void)out_size; (void)ws_size;

  const float* node_feats = (const float*)d_in[0];
  const float* edge_feats = (const float*)d_in[1];
  const int*   src        = (const int*)d_in[2];
  const int*   dst        = (const int*)d_in[3];
  const float* edge_w1    = (const float*)d_in[4];
  const float* edge_b1    = (const float*)d_in[5];
  const float* edge_w2    = (const float*)d_in[6];
  const float* edge_b2    = (const float*)d_in[7];
  const float* edge_ln_s  = (const float*)d_in[8];
  const float* edge_ln_b  = (const float*)d_in[9];
  const float* node_w1    = (const float*)d_in[10];
  const float* node_b1    = (const float*)d_in[11];
  const float* node_w2    = (const float*)d_in[12];
  const float* node_b2    = (const float*)d_in[13];
  const float* node_ln_s  = (const float*)d_in[14];
  const float* node_ln_b  = (const float*)d_in[15];

  // residual state lives in d_out: [n (10242x512) | e (81920x512)] f32
  float* n_work = (float*)d_out;
  float* e_work = n_work + (size_t)NNODES * HDIM;

  // workspace: bf16 transposed weights | agg f32 | h bf16
  char* ws = (char*)d_ws;
  const size_t LW = (size_t)7 * HDIM * HDIM;  // bf16 elems per layer (3HH+HH+2HH+HH)
  __bf16* wbf = (__bf16*)ws;
  const size_t OFF_AGG = (size_t)NLAYERS * LW * sizeof(__bf16);
  float* agg = (float*)(ws + OFF_AGG);
  const size_t OFF_H = OFF_AGG + (size_t)NNODES * HDIM * sizeof(float);
  __bf16* hbuf = (__bf16*)(ws + OFF_H);

  (void)hipMemcpyAsync(n_work, node_feats, (size_t)NNODES * HDIM * sizeof(float),
                       hipMemcpyDeviceToDevice, stream);
  (void)hipMemcpyAsync(e_work, edge_feats, (size_t)NEDGES * HDIM * sizeof(float),
                       hipMemcpyDeviceToDevice, stream);

  // one-time weight transpose + bf16 convert
  for (int l = 0; l < NLAYERS; ++l) {
    __bf16* ew1t = wbf + (size_t)l * LW;
    __bf16* ew2t = ew1t + (size_t)3 * HDIM * HDIM;
    __bf16* nw1t = ew2t + (size_t)HDIM * HDIM;
    __bf16* nw2t = nw1t + (size_t)2 * HDIM * HDIM;
    auto tl = [&](const float* w, __bf16* wt, int K, int N) {
      const size_t e = (size_t)K * N;
      wtrans_kernel<<<(unsigned)((e + 255) / 256), 256, 0, stream>>>(w, wt, K, N);
    };
    tl(edge_w1 + (size_t)l * 3 * HDIM * HDIM, ew1t, 3 * HDIM, HDIM);
    tl(edge_w2 + (size_t)l * HDIM * HDIM,     ew2t, HDIM,     HDIM);
    tl(node_w1 + (size_t)l * 2 * HDIM * HDIM, nw1t, 2 * HDIM, HDIM);
    tl(node_w2 + (size_t)l * HDIM * HDIM,     nw2t, HDIM,     HDIM);
  }

  for (int l = 0; l < NLAYERS; ++l) {
    __bf16* ew1t = wbf + (size_t)l * LW;
    __bf16* ew2t = ew1t + (size_t)3 * HDIM * HDIM;
    __bf16* nw1t = ew2t + (size_t)HDIM * HDIM;
    __bf16* nw2t = nw1t + (size_t)2 * HDIM * HDIM;

    // edge MLP layer 1: gather-concat fused into A-tile load, SiLU epilogue -> bf16 h
    gemm_fused<1, 0, 3 * HDIM><<<NEDGES / BM, 256, 0, stream>>>(
        nullptr, e_work, n_work, n_work, src, dst,
        ew1t, NEDGES,
        edge_b1 + (size_t)l * HDIM, hbuf,
        nullptr, nullptr, nullptr);

    // edge MLP layer 2: bias + LN + residual fused -> e (in place)
    gemm_fused<0, 1, HDIM><<<NEDGES / BM, 256, 0, stream>>>(
        hbuf, nullptr, nullptr, nullptr, nullptr, nullptr,
        ew2t, NEDGES,
        edge_b2 + (size_t)l * HDIM, nullptr,
        e_work, edge_ln_s + (size_t)l * HDIM, edge_ln_b + (size_t)l * HDIM);

    // aggregate edges to nodes
    zero_kernel<<<(NNODES * HDIM + 255) / 256, 256, 0, stream>>>(agg, NNODES * HDIM);
    scatter_kernel<<<NEDGES, 256, 0, stream>>>(e_work, dst, agg);

    // node MLP layer 1: concat(n, agg) fused into A-tile load
    gemm_fused<2, 0, 2 * HDIM><<<(NNODES + BM - 1) / BM, 256, 0, stream>>>(
        nullptr, n_work, agg, nullptr, nullptr, nullptr,
        nw1t, NNODES,
        node_b1 + (size_t)l * HDIM, hbuf,
        nullptr, nullptr, nullptr);

    // node MLP layer 2: bias + LN + residual -> n (in place)
    gemm_fused<0, 1, HDIM><<<(NNODES + BM - 1) / BM, 256, 0, stream>>>(
        hbuf, nullptr, nullptr, nullptr, nullptr, nullptr,
        nw2t, NNODES,
        node_b2 + (size_t)l * HDIM, nullptr,
        n_work, node_ln_s + (size_t)l * HDIM, node_ln_b + (size_t)l * HDIM);
  }
}
